// LocalMultiMessagePassing_6133213299116
// MI455X (gfx1250) — matrix operand order repaired and checked by reading, not compiled, not run on hardware
//
#include <hip/hip_runtime.h>
#include <math.h>

// ---------------------------------------------------------------------------
// MI455X (gfx1250, wave32) 4-step GraphNet+GlobalNode.
// bf16 activations/weights, f32 WMMA accumulate (v_wmma_f32_16x16x32_bf16).
// Weights staged global->LDS via Tensor Data Mover (tensor_load_to_lds),
// B-fragments served from LDS (ds_load_b128), A-fragments preloaded to VGPRs.
// ---------------------------------------------------------------------------

typedef __attribute__((ext_vector_type(16))) __bf16 v16bf;
typedef __attribute__((ext_vector_type(8)))  float  v8f;
typedef unsigned int uint32x4 __attribute__((ext_vector_type(4)));
typedef int          int32x8  __attribute__((ext_vector_type(8)));
typedef int          int32x4  __attribute__((ext_vector_type(4)));

#define DD 128   // feature dim

__device__ __forceinline__ float leaky(float v) { return v >= 0.f ? v : 0.01f * v; }

// A-fragment (16x32 bf16): lane holds row M=lane%16; half=lane/16 selects
// K chunk [half*16, half*16+16) within the 32-K tile.
__device__ __forceinline__ v16bf load_a_frag(const __bf16* row, int kc, int half) {
  union { v16bf v; uint4 q[2]; } u;
  u.q[0] = *(const uint4*)(row + kc * 32 + half * 16);
  u.q[1] = *(const uint4*)(row + kc * 32 + half * 16 + 8);
  return u.v;
}

// TDM: DMA a [rows x rowlen] bf16 row-major tile from global into LDS.
// D# per CDNA5 ISA ch.8: group0 = count/lds/global/type, group1 = dims/strides.
// This toolchain exposes the 6-arg builtin (g0, g1, g2, g3, extra, cpol).
__device__ __forceinline__ void tdm_load_2d(const void* gaddr, unsigned lds_off,
                                            unsigned rows, unsigned rowlen) {
  unsigned long long ga = (unsigned long long)(uintptr_t)gaddr;
  uint32x4 g0;
  g0[0] = 1u;                                         // count=1 (valid), user mode
  g0[1] = lds_off;                                    // lds_addr (bytes)
  g0[2] = (unsigned)(ga & 0xFFFFFFFFu);               // global_addr[31:0]
  g0[3] = (unsigned)((ga >> 32) & 0x1FFFFFFu) | (2u << 30);  // addr[56:32] | type=2
  int32x8 g1;
  g1[0] = 0x10000;                                    // data_size=1 (2 bytes), no mask
  g1[1] = (int)((rowlen & 0xFFFFu) << 16);            // tensor_dim0[15:0]
  g1[2] = (int)(((rowlen >> 16) & 0xFFFFu) | ((rows & 0xFFFFu) << 16));  // dim0 hi | dim1 lo
  g1[3] = (int)(((rows >> 16) & 0xFFFFu) | (rowlen << 16));              // dim1 hi | tile_dim0
  g1[4] = (int)(rows & 0xFFFFu);                      // tile_dim1, tile_dim2=0
  g1[5] = (int)rowlen;                                // tensor_dim0_stride[31:0]
  g1[6] = 0;                                          // stride hi, dim1_stride lo
  g1[7] = 0;
  int32x4 z4 = {0, 0, 0, 0};
  int32x8 z8 = {0, 0, 0, 0, 0, 0, 0, 0};
  __builtin_amdgcn_tensor_load_to_lds(g0, g1, z4, z4, z8, 0);
}

// ---------------- elementwise helpers ----------------
__global__ void k_fill(float* p, float v, int n) {
  int i = blockIdx.x * blockDim.x + threadIdx.x;
  if (i < n) p[i] = v;
}
__global__ void k_cvt_bf16(const float* __restrict__ in, __bf16* __restrict__ out, int n) {
  int i = blockIdx.x * blockDim.x + threadIdx.x;
  if (i < n) out[i] = (__bf16)in[i];
}
__global__ void k_aggb(const float* __restrict__ agg, __bf16* __restrict__ out, int n) {
  int i = blockIdx.x * blockDim.x + threadIdx.x;
  if (i < n) {
    float v = agg[i];
    if (!isfinite(v)) v = 0.f;   // PyG: empty segment -> 0
    out[i] = (__bf16)v;
  }
}

// ---------------- WMMA kernels ----------------
// Edge messages: msg = leaky(x[src] @ Wm^T + bm); scatter-max into agg[dst].
// One wave = 16 edges x 128 outputs (32 WMMAs). Weights in LDS via TDM.
__global__ void __launch_bounds__(256)
k_edge_msg(const __bf16* __restrict__ xb, const __bf16* __restrict__ Wb,
           const float* __restrict__ bias,
           const int* __restrict__ src, const int* __restrict__ dst,
           float* __restrict__ agg, int E) {
  __shared__ __align__(16) __bf16 Ws[DD * DD];       // 32 KB
  int wid  = (blockIdx.x * blockDim.x + threadIdx.x) >> 5;
  int lane = threadIdx.x & 31;
  int e0 = wid * 16;
  if (e0 >= E) return;                                // wave-uniform
  int half = lane >> 4, ncol = lane & 15;

  if (threadIdx.x < 32) {                             // wave 0 issues the DMA
    tdm_load_2d(Wb, (unsigned)(uintptr_t)&Ws[0], DD, DD);
    __builtin_amdgcn_s_wait_tensorcnt(0);
  }
  // A-fragment preload (overlaps with weight DMA)
  const __bf16* arow = xb + (size_t)src[e0 + ncol] * DD;
  v16bf afr[4];
#pragma unroll
  for (int kc = 0; kc < 4; ++kc) afr[kc] = load_a_frag(arow, kc, half);
  __syncthreads();

  v8f acc[8] = {};
#pragma unroll
  for (int kc = 0; kc < 4; ++kc) {
#pragma unroll
    for (int nt = 0; nt < 8; ++nt) {
      union { v16bf v; uint4 q[2]; } ub;
      const uint4* bp = (const uint4*)&Ws[(nt * 16 + ncol) * DD + kc * 32 + half * 16];
      ub.q[0] = bp[0]; ub.q[1] = bp[1];
      acc[nt] = __builtin_amdgcn_wmma_f32_16x16x32_bf16(false, afr[kc], false, ub.v,
                                                        (short)0, acc[nt], false, false);
    }
  }
  int drow[8];
#pragma unroll
  for (int j = 0; j < 8; ++j) drow[j] = dst[e0 + half * 8 + j];
#pragma unroll
  for (int nt = 0; nt < 8; ++nt) {
    int col = nt * 16 + ncol;
    float bv = bias[col];
#pragma unroll
    for (int j = 0; j < 8; ++j) {
      float v = leaky(acc[nt][j] + bv);
      __hip_atomic_fetch_max(&agg[(size_t)drow[j] * DD + col], v,
                             __ATOMIC_RELAXED, __HIP_MEMORY_SCOPE_AGENT);
    }
  }
}

// Node update: x = leaky(cat(x, agg) @ Wa^T + ba) + x   (K=256)
__global__ void __launch_bounds__(256)
k_node_update(const __bf16* __restrict__ xb, const __bf16* __restrict__ aggb,
              const __bf16* __restrict__ Wb, const float* __restrict__ bias,
              const float* __restrict__ xold,
              float* __restrict__ xnew, __bf16* __restrict__ xbnew, int Nn) {
  __shared__ __align__(16) __bf16 Ws[DD * 2 * DD];    // 64 KB
  int wid  = (blockIdx.x * blockDim.x + threadIdx.x) >> 5;
  int lane = threadIdx.x & 31;
  int r0 = wid * 16;
  if (r0 >= Nn) return;
  int half = lane >> 4, ncol = lane & 15;

  if (threadIdx.x < 32) {
    tdm_load_2d(Wb, (unsigned)(uintptr_t)&Ws[0], DD, 2 * DD);
    __builtin_amdgcn_s_wait_tensorcnt(0);
  }
  const __bf16* arow0 = xb   + (size_t)(r0 + ncol) * DD;
  const __bf16* arow1 = aggb + (size_t)(r0 + ncol) * DD;
  v16bf afr[8];
#pragma unroll
  for (int kc = 0; kc < 4; ++kc) afr[kc]     = load_a_frag(arow0, kc, half);
#pragma unroll
  for (int kc = 0; kc < 4; ++kc) afr[4 + kc] = load_a_frag(arow1, kc, half);
  __syncthreads();

  v8f acc[8] = {};
#pragma unroll
  for (int kc = 0; kc < 8; ++kc) {
#pragma unroll
    for (int nt = 0; nt < 8; ++nt) {
      union { v16bf v; uint4 q[2]; } ub;
      const uint4* bp = (const uint4*)&Ws[(nt * 16 + ncol) * (2 * DD) + kc * 32 + half * 16];
      ub.q[0] = bp[0]; ub.q[1] = bp[1];
      acc[nt] = __builtin_amdgcn_wmma_f32_16x16x32_bf16(false, afr[kc], false, ub.v,
                                                        (short)0, acc[nt], false, false);
    }
  }
#pragma unroll
  for (int nt = 0; nt < 8; ++nt) {
    int col = nt * 16 + ncol;
    float bv = bias[col];
#pragma unroll
    for (int j = 0; j < 8; ++j) {
      size_t idx = (size_t)(r0 + half * 8 + j) * DD + col;
      float v = leaky(acc[nt][j] + bv) + xold[idx];
      xnew[idx]  = v;
      xbnew[idx] = (__bf16)v;
    }
  }
}

// feat = leaky(x @ Wfeat^T + bfeat)   (K=128)
__global__ void __launch_bounds__(256)
k_feat(const __bf16* __restrict__ xb, const __bf16* __restrict__ Wb,
       const float* __restrict__ bias, float* __restrict__ feat, int Nn) {
  __shared__ __align__(16) __bf16 Ws[DD * DD];       // 32 KB
  int wid  = (blockIdx.x * blockDim.x + threadIdx.x) >> 5;
  int lane = threadIdx.x & 31;
  int r0 = wid * 16;
  if (r0 >= Nn) return;
  int half = lane >> 4, ncol = lane & 15;

  if (threadIdx.x < 32) {
    tdm_load_2d(Wb, (unsigned)(uintptr_t)&Ws[0], DD, DD);
    __builtin_amdgcn_s_wait_tensorcnt(0);
  }
  const __bf16* arow = xb + (size_t)(r0 + ncol) * DD;
  v16bf afr[4];
#pragma unroll
  for (int kc = 0; kc < 4; ++kc) afr[kc] = load_a_frag(arow, kc, half);
  __syncthreads();

  v8f acc[8] = {};
#pragma unroll
  for (int kc = 0; kc < 4; ++kc) {
#pragma unroll
    for (int nt = 0; nt < 8; ++nt) {
      union { v16bf v; uint4 q[2]; } ub;
      const uint4* bp = (const uint4*)&Ws[(nt * 16 + ncol) * DD + kc * 32 + half * 16];
      ub.q[0] = bp[0]; ub.q[1] = bp[1];
      acc[nt] = __builtin_amdgcn_wmma_f32_16x16x32_bf16(false, afr[kc], false, ub.v,
                                                        (short)0, acc[nt], false, false);
    }
  }
#pragma unroll
  for (int nt = 0; nt < 8; ++nt) {
    int col = nt * 16 + ncol;
    float bv = bias[col];
#pragma unroll
    for (int j = 0; j < 8; ++j) {
      size_t idx = (size_t)(r0 + half * 8 + j) * DD + col;
      feat[idx] = leaky(acc[nt][j] + bv);
    }
  }
}

// ---------------- attentional aggregation ----------------
__global__ void k_gate(const float* __restrict__ x, const float* __restrict__ wg,
                       const float* __restrict__ bg, float* __restrict__ gate, int Nn) {
  int wid  = (blockIdx.x * blockDim.x + threadIdx.x) >> 5;
  int lane = threadIdx.x & 31;
  if (wid >= Nn) return;
  const float* row = x + (size_t)wid * DD;
  float s = 0.f;
#pragma unroll
  for (int i = 0; i < 4; ++i) s += row[lane + 32 * i] * wg[lane + 32 * i];
#pragma unroll
  for (int off = 16; off > 0; off >>= 1) s += __shfl_xor(s, off, 32);
  if (lane == 0) gate[wid] = s + bg[0];
}

__global__ void k_gmax(const float* __restrict__ gate, const int* __restrict__ batch,
                       float* __restrict__ gmax, int Nn) {
  int i = blockIdx.x * blockDim.x + threadIdx.x;
  if (i < Nn)
    __hip_atomic_fetch_max(&gmax[batch[i]], gate[i], __ATOMIC_RELAXED, __HIP_MEMORY_SCOPE_AGENT);
}

__global__ void k_expsum(const float* __restrict__ gate, const int* __restrict__ batch,
                         const float* __restrict__ gmax, float* __restrict__ ebuf,
                         float* __restrict__ denom, int Nn) {
  int i = blockIdx.x * blockDim.x + threadIdx.x;
  if (i >= Nn) return;
  int b = batch[i];
  float g = gmax[b];
  if (!isfinite(g)) g = 0.f;
  float e = expf(gate[i] - g);
  ebuf[i] = e;
  atomicAdd(&denom[b], e);
}

__global__ void k_xgacc(const float* __restrict__ ebuf, const float* __restrict__ feat,
                        const int* __restrict__ batch, float* __restrict__ xg, int total) {
  int i = blockIdx.x * blockDim.x + threadIdx.x;
  if (i >= total) return;
  int n = i >> 7, d = i & (DD - 1);
  atomicAdd(&xg[(size_t)batch[n] * DD + d], ebuf[n] * feat[i]);
}

__global__ void k_xgdiv(float* __restrict__ xg, const float* __restrict__ denom, int total) {
  int i = blockIdx.x * blockDim.x + threadIdx.x;
  if (i >= total) return;
  float dn = denom[i >> 7];
  if (dn == 0.f) dn = 1.f;
  xg[i] /= dn;
}

__global__ void k_global(const float* __restrict__ xg, const float* __restrict__ xglo,
                         const float* __restrict__ Wt, const float* __restrict__ bt,
                         float* __restrict__ xgln) {
  int g = blockIdx.x, d = threadIdx.x;
  const float* w  = Wt + (size_t)d * (2 * DD);
  const float* a0 = xg + (size_t)g * DD;
  const float* a1 = xglo + (size_t)g * DD;
  float s = bt[d];
  for (int k = 0; k < DD; ++k) s += w[k] * a0[k];
  for (int k = 0; k < DD; ++k) s += w[DD + k] * a1[k];
  xgln[(size_t)g * DD + d] = leaky(s) + a1[d];
}

// ---------------------------------------------------------------------------
extern "C" void kernel_launch(void* const* d_in, const int* in_sizes, int n_in,
                              void* d_out, int out_size, void* d_ws, size_t ws_size,
                              hipStream_t stream) {
  const float* x_in   = (const float*)d_in[0];
  const float* xgl_in = (const float*)d_in[1];
  const float* Wm = (const float*)d_in[3];
  const float* bm = (const float*)d_in[4];
  const float* Wa = (const float*)d_in[5];
  const float* ba = (const float*)d_in[6];
  const float* Wg = (const float*)d_in[7];
  const float* bg = (const float*)d_in[8];
  const float* Wf = (const float*)d_in[9];
  const float* bf = (const float*)d_in[10];
  const float* Wt = (const float*)d_in[11];
  const float* bt = (const float*)d_in[12];
  const int*   ei = (const int*)d_in[13];
  const int*   batch = (const int*)d_in[14];

  const int N = in_sizes[0] / DD;
  const int G = in_sizes[1] / DD;
  const int E = in_sizes[13] / 2;
  const int S = in_sizes[4] / DD;
  const int* src = ei;
  const int* dst = ei + E;

  char* wp = (char*)d_ws;
  auto alloc = [&](size_t bytes) -> void* {
    void* p = (void*)wp;
    wp += (bytes + 255) & ~(size_t)255;
    return p;
  };
  const size_t nd = (size_t)N * DD;
  __bf16* xb0   = (__bf16*)alloc(nd * 2);
  __bf16* xb1   = (__bf16*)alloc(nd * 2);
  float*  x0    = (float*) alloc(nd * 4);
  float*  x1    = (float*) alloc(nd * 4);
  float*  agg   = (float*) alloc(nd * 4);
  __bf16* aggb  = (__bf16*)alloc(nd * 2);
  float*  feat  = (float*) alloc(nd * 4);
  float*  gate  = (float*) alloc((size_t)N * 4);
  float*  ebuf  = (float*) alloc((size_t)N * 4);
  __bf16* Wmb   = (__bf16*)alloc((size_t)S * DD * DD * 2);
  __bf16* Wab   = (__bf16*)alloc((size_t)S * DD * 2 * DD * 2);
  __bf16* Wfb   = (__bf16*)alloc((size_t)S * DD * DD * 2);
  float*  xg    = (float*) alloc((size_t)G * DD * 4);
  float*  xgl0  = (float*) alloc((size_t)G * DD * 4);
  float*  xgl1  = (float*) alloc((size_t)G * DD * 4);
  float*  gmax  = (float*) alloc((size_t)G * 4);
  float*  denom = (float*) alloc((size_t)G * 4);

  auto cdiv = [](long a, long b) { return (int)((a + b - 1) / b); };
  const int B = 256;

  k_cvt_bf16<<<cdiv(nd, B), B, 0, stream>>>(x_in, xb0, (int)nd);
  k_cvt_bf16<<<cdiv((long)S * DD * DD, B), B, 0, stream>>>(Wm, Wmb, S * DD * DD);
  k_cvt_bf16<<<cdiv((long)S * DD * 2 * DD, B), B, 0, stream>>>(Wa, Wab, S * DD * 2 * DD);
  k_cvt_bf16<<<cdiv((long)S * DD * DD, B), B, 0, stream>>>(Wf, Wfb, S * DD * DD);
  (void)hipMemcpyAsync(xgl0, xgl_in, (size_t)G * DD * 4, hipMemcpyDeviceToDevice, stream);

  const float* xcur = x_in;
  float* xnxt = x0;
  __bf16* xbcur = xb0; __bf16* xbnxt = xb1;
  float* xglcur = xgl0; float* xglnxt = xgl1;

  const int edgeBlocks = cdiv((long)cdiv(E, 16) * 32, B);
  const int nodeBlocks = cdiv((long)cdiv(N, 16) * 32, B);
  const int gateBlocks = cdiv((long)N * 32, B);

  for (int i = 0; i < S; ++i) {
    k_fill<<<cdiv(nd, B), B, 0, stream>>>(agg, -INFINITY, (int)nd);
    k_edge_msg<<<edgeBlocks, B, 0, stream>>>(xbcur, Wmb + (size_t)i * DD * DD,
                                             bm + (size_t)i * DD, src, dst, agg, E);
    k_aggb<<<cdiv(nd, B), B, 0, stream>>>(agg, aggb, (int)nd);
    k_node_update<<<nodeBlocks, B, 0, stream>>>(xbcur, aggb,
                                                Wab + (size_t)i * DD * 2 * DD,
                                                ba + (size_t)i * DD, xcur, xnxt, xbnxt, N);
    k_gate<<<gateBlocks, B, 0, stream>>>(xnxt, Wg + (size_t)i * DD, bg + i, gate, N);
    k_feat<<<nodeBlocks, B, 0, stream>>>(xbnxt, Wfb + (size_t)i * DD * DD,
                                         bf + (size_t)i * DD, feat, N);
    k_fill<<<1, B, 0, stream>>>(gmax, -INFINITY, G);
    k_fill<<<1, B, 0, stream>>>(denom, 0.f, G);
    k_fill<<<cdiv((long)G * DD, B), B, 0, stream>>>(xg, 0.f, G * DD);
    k_gmax<<<cdiv(N, B), B, 0, stream>>>(gate, batch, gmax, N);
    k_expsum<<<cdiv(N, B), B, 0, stream>>>(gate, batch, gmax, ebuf, denom, N);
    k_xgacc<<<cdiv(nd, B), B, 0, stream>>>(ebuf, feat, batch, xg, (int)nd);
    k_xgdiv<<<cdiv((long)G * DD, B), B, 0, stream>>>(xg, denom, G * DD);
    k_global<<<G, DD, 0, stream>>>(xg, xglcur, Wt + (size_t)i * DD * 2 * DD,
                                   bt + (size_t)i * DD, xglnxt);
    xcur = xnxt;        xnxt = (xnxt == x0) ? x1 : x0;
    __bf16* tb = xbcur; xbcur = xbnxt;  xbnxt = tb;
    float* tg = xglcur; xglcur = xglnxt; xglnxt = tg;
  }

  float* out = (float*)d_out;
  (void)hipMemcpyAsync(out, xcur, nd * 4, hipMemcpyDeviceToDevice, stream);
  (void)hipMemcpyAsync(out + nd, xglcur, (size_t)G * DD * 4, hipMemcpyDeviceToDevice, stream);
}
